// Pos_emb_7181185319292
// MI455X (gfx1250) — matrix operand order repaired and checked by reading, not compiled, not run on hardware
//
#include <hip/hip_runtime.h>
#include <hip/hip_bf16.h>
#include <math.h>

// ---------------------------------------------------------------------------
// MI455X (gfx1250) transformer forward:  wave32, WMMA bf16 (f32 accumulate),
// async global->LDS weight staging (ASYNCcnt) in the GEMM hot loop.
// ---------------------------------------------------------------------------
typedef unsigned short bf16_t;
typedef __attribute__((ext_vector_type(16))) __bf16 v16bf;
typedef __attribute__((ext_vector_type(8)))  float  v8f;

struct Pack32 { uint4 a, b; };

__device__ __forceinline__ v16bf pack_frag(uint4 lo, uint4 hi) {
  Pack32 p; p.a = lo; p.b = hi;
  return __builtin_bit_cast(v16bf, p);
}

__device__ __forceinline__ bf16_t f2bf(float f) {
  unsigned u = __builtin_bit_cast(unsigned, f);
  u += 0x7FFFu + ((u >> 16) & 1u);           // round-to-nearest-even
  return (bf16_t)(u >> 16);
}

// CDNA5 async copy: 16B per lane, global -> LDS, tracked by ASYNCcnt.
__device__ __forceinline__ void async_copy16(unsigned lds_off, const void* gptr) {
  asm volatile("global_load_async_to_lds_b128 %0, %1, off"
               :: "v"(lds_off), "v"(gptr) : "memory");
}
__device__ __forceinline__ void wait_async0() {
  asm volatile("s_wait_asynccnt 0x0" ::: "memory");
}

// A fragment (16x32 bf16): lane L -> row = L&15; K = {0..7,16..23} (+8 upper half)
__device__ __forceinline__ v16bf load_afrag(const bf16_t* __restrict__ A,
                                            size_t row, int lda, int k0, int lane) {
  const bf16_t* p = A + row * (size_t)lda + k0 + ((lane & 16) ? 8 : 0);
  uint4 lo = *(const uint4*)p;
  uint4 hi = *(const uint4*)(p + 16);
  return pack_frag(lo, hi);
}

// ---------------------------------------------------------------------------
// Generic WMMA GEMM:  C[M,N] = A[M,K](bf16) * W[N,K]^T(bf16) + bias
// 256 threads = 8 waves; block tile 128 x (NT*16); wave = 16 rows x NT tiles.
// Weight slice (BN x 32) double-buffered in LDS via async global->LDS loads.
// EPI 0: store bf16          (QKV)
// EPI 1: ReLU, store bf16    (FF1)
// EPI 2: + residual(Xf), LayerNorm row (N == NT*16), write Xf(f32)+Xb(bf16)
// ---------------------------------------------------------------------------
template<int NT, int EPI>
__global__ __launch_bounds__(256) void gemm_kernel(
    const bf16_t* __restrict__ A, const bf16_t* __restrict__ W,
    const float* __restrict__ bias, int N, int K,
    bf16_t* __restrict__ Obf,
    const float* __restrict__ lns, const float* __restrict__ lnb,
    float* __restrict__ Xf, bf16_t* __restrict__ Xb)
{
  constexpr int BN = NT * 16;
  __shared__ bf16_t Ws[2][BN][40];   // 32 data + 8 pad per row (80B rows)

  const int tid  = threadIdx.x;
  const int lane = tid & 31;
  const int wave = tid >> 5;
  const int col  = lane & 15;
  const int half = (lane >> 4) & 1;
  const int m_base = blockIdx.y * 128 + wave * 16;
  const int n0 = blockIdx.x * BN;

  // Stage one BN x 32 weight slice into LDS buffer `buf` (async, 16B/lane).
  auto stage = [&](int buf, int k0) {
#pragma unroll
    for (int rr = 0; rr < NT / 4; ++rr) {
      const int i   = rr * 256 + tid;
      const int row = i >> 2;        // 4 x 16B chunks per 64B row
      const int ch  = i & 3;
      unsigned lds = (unsigned)(uintptr_t)&Ws[buf][row][ch * 8];
      const bf16_t* g = W + (size_t)(n0 + row) * K + k0 + ch * 8;
      async_copy16(lds, g);
    }
  };

  v8f acc[NT];
#pragma unroll
  for (int t = 0; t < NT; ++t) acc[t] = (v8f){};

  stage(0, 0);
  wait_async0();
  __syncthreads();

  int buf = 0;
  for (int k0 = 0; k0 < K; k0 += 32) {
    const bool more = (k0 + 32) < K;
    if (more) stage(buf ^ 1, k0 + 32);   // overlap with this k-step's WMMAs

    v16bf af = load_afrag(A, (size_t)(m_base + col), K, k0, lane);
#pragma unroll
    for (int t = 0; t < NT; ++t) {
      const bf16_t* bp = &Ws[buf][t * 16 + col][half * 8];
      v16bf bfr = pack_frag(*(const uint4*)bp, *(const uint4*)(bp + 16));
      acc[t] = __builtin_amdgcn_wmma_f32_16x16x32_bf16(
          false, af, false, bfr, (short)0, acc[t], false, false);
    }

    if (more) {
      wait_async0();      // own async loads landed
      __syncthreads();    // everyone's loads landed, everyone done reading buf
      buf ^= 1;
    }
  }

  if constexpr (EPI == 0 || EPI == 1) {
#pragma unroll
    for (int t = 0; t < NT; ++t) {
      const int c = n0 + t * 16 + col;
      const float bv = bias[c];
#pragma unroll
      for (int r = 0; r < 8; ++r) {
        const int m = m_base + r + half * 8;
        float v = acc[t][r] + bv;
        if (EPI == 1) v = fmaxf(v, 0.0f);
        Obf[(size_t)m * N + c] = f2bf(v);
      }
    }
  } else {
    // bias + residual, then in-register LayerNorm (wave owns whole rows).
    float g[NT], bb[NT];
#pragma unroll
    for (int t = 0; t < NT; ++t) {
      const int c = n0 + t * 16 + col;
      const float bv = bias[c];
      g[t] = lns[c]; bb[t] = lnb[c];
#pragma unroll
      for (int r = 0; r < 8; ++r) {
        const int m = m_base + r + half * 8;
        acc[t][r] += bv + Xf[(size_t)m * N + c];
      }
    }
    const float invN = 1.0f / (float)(NT * 16);
#pragma unroll
    for (int r = 0; r < 8; ++r) {
      float s = 0.f;
#pragma unroll
      for (int t = 0; t < NT; ++t) s += acc[t][r];
      s += __shfl_xor(s, 1); s += __shfl_xor(s, 2);
      s += __shfl_xor(s, 4); s += __shfl_xor(s, 8);
      const float mu = s * invN;
      float vs = 0.f;
#pragma unroll
      for (int t = 0; t < NT; ++t) { float d = acc[t][r] - mu; vs += d * d; }
      vs += __shfl_xor(vs, 1); vs += __shfl_xor(vs, 2);
      vs += __shfl_xor(vs, 4); vs += __shfl_xor(vs, 8);
      const float inv = rsqrtf(vs * invN + 1e-5f);
      const int m = m_base + r + half * 8;
#pragma unroll
      for (int t = 0; t < NT; ++t) {
        const int c = n0 + t * 16 + col;
        const float y = (acc[t][r] - mu) * inv * g[t] + bb[t];
        Xf[(size_t)m * N + c] = y;
        Xb[(size_t)m * N + c] = f2bf(y);
      }
    }
  }
}

// ---------------------------------------------------------------------------
// Attention: one block per (128-row strip, head h, batch n).
// Rows = 't' axis (256 long). qkv row index = t*129 + n, width 768.
// ---------------------------------------------------------------------------
__global__ __launch_bounds__(256) void attn_kernel(
    const bf16_t* __restrict__ qkv,  // [33024][768]
    bf16_t* __restrict__ O)          // [33024][256]
{
  __shared__ bf16_t Qs[128][40];     // rows padded to 80B
  __shared__ bf16_t Ks[256][40];
  __shared__ bf16_t Vt[32][264];     // V transposed: [dh][s]
  __shared__ bf16_t Ps[128][264];    // softmax probs

  const int tid  = threadIdx.x;
  const int lane = tid & 31;
  const int wave = tid >> 5;
  const int col  = lane & 15;
  const int half = (lane >> 4) & 1;
  const int mblk = blockIdx.x;       // 0..1
  const int h    = blockIdx.y;       // 0..7
  const int n    = blockIdx.z;       // 0..128
  const int t0   = mblk * 128;

  { // Q strip (128 x 32)
    const int tt  = tid >> 1;
    const int dh0 = (tid & 1) * 16;
    const bf16_t* src = qkv + ((size_t)(t0 + tt) * 129 + n) * 768 + h * 32 + dh0;
    *(uint4*)&Qs[tt][dh0]     = *(const uint4*)src;
    *(uint4*)&Qs[tt][dh0 + 8] = *(const uint4*)(src + 8);
  }
  { // K (256 x 32)
    const int s = tid;
    const bf16_t* src = qkv + ((size_t)s * 129 + n) * 768 + 256 + h * 32;
    *(uint4*)&Ks[s][0]  = ((const uint4*)src)[0];
    *(uint4*)&Ks[s][8]  = ((const uint4*)src)[1];
    *(uint4*)&Ks[s][16] = ((const uint4*)src)[2];
    *(uint4*)&Ks[s][24] = ((const uint4*)src)[3];
  }
  { // V transposed (32 x 256)
    const int s = tid;
    const bf16_t* src = qkv + ((size_t)s * 129 + n) * 768 + 512 + h * 32;
    uint4 q[4] = { ((const uint4*)src)[0], ((const uint4*)src)[1],
                   ((const uint4*)src)[2], ((const uint4*)src)[3] };
#pragma unroll
    for (int d = 0; d < 32; ++d) Vt[d][s] = ((const bf16_t*)q)[d];
  }
  __syncthreads();

  // ---- scores: S = Q K^T  (wave: 16 rows x 256 cols, K=32) ----
  v8f sa[16];
  {
    const int row = wave * 16 + col;
    const bf16_t* p = &Qs[row][half * 8];
    v16bf af = pack_frag(*(const uint4*)p, *(const uint4*)(p + 16));
#pragma unroll
    for (int j = 0; j < 16; ++j) {
      const bf16_t* bp = &Ks[j * 16 + col][half * 8];
      v16bf bfr = pack_frag(*(const uint4*)bp, *(const uint4*)(bp + 16));
      v8f z = (v8f){};
      sa[j] = __builtin_amdgcn_wmma_f32_16x16x32_bf16(
          false, af, false, bfr, (short)0, z, false, false);
    }
  }

  // ---- softmax over s (full row in this wave's registers) ----
  const float scale = 0.17677669529663687f; // 1/sqrt(32)
#pragma unroll
  for (int r = 0; r < 8; ++r) {
    float mx = -3.0e38f;
#pragma unroll
    for (int j = 0; j < 16; ++j) mx = fmaxf(mx, sa[j][r]);
    mx = fmaxf(mx, __shfl_xor(mx, 1)); mx = fmaxf(mx, __shfl_xor(mx, 2));
    mx = fmaxf(mx, __shfl_xor(mx, 4)); mx = fmaxf(mx, __shfl_xor(mx, 8));
    float sum = 0.f;
#pragma unroll
    for (int j = 0; j < 16; ++j) {
      float e = __expf((sa[j][r] - mx) * scale);
      sa[j][r] = e; sum += e;
    }
    sum += __shfl_xor(sum, 1); sum += __shfl_xor(sum, 2);
    sum += __shfl_xor(sum, 4); sum += __shfl_xor(sum, 8);
    const float inv = 1.0f / sum;
    const int prow = wave * 16 + r + half * 8;
#pragma unroll
    for (int j = 0; j < 16; ++j) Ps[prow][j * 16 + col] = f2bf(sa[j][r] * inv);
  }
  __syncthreads();

  // ---- O = P V  (wave: 16 rows x 32 cols, K=256) ----
  v8f oa[2]; oa[0] = (v8f){}; oa[1] = (v8f){};
#pragma unroll
  for (int ks = 0; ks < 8; ++ks) {
    const bf16_t* ap = &Ps[wave * 16 + col][ks * 32 + half * 8];
    v16bf af = pack_frag(*(const uint4*)ap, *(const uint4*)(ap + 16));
#pragma unroll
    for (int nt = 0; nt < 2; ++nt) {
      const bf16_t* bp = &Vt[nt * 16 + col][ks * 32 + half * 8];
      v16bf bfr = pack_frag(*(const uint4*)bp, *(const uint4*)(bp + 16));
      oa[nt] = __builtin_amdgcn_wmma_f32_16x16x32_bf16(
          false, af, false, bfr, (short)0, oa[nt], false, false);
    }
  }
#pragma unroll
  for (int nt = 0; nt < 2; ++nt) {
#pragma unroll
    for (int r = 0; r < 8; ++r) {
      const int t = t0 + wave * 16 + r + half * 8;
      const int c = h * 32 + nt * 16 + col;
      O[((size_t)t * 129 + n) * 256 + c] = f2bf(oa[nt][r]);
    }
  }
}

// ---------------------------------------------------------------------------
// Embedding: x[b, 0] = emb[1]; x[b, m+1] = covered ? emb[pos2[b,idx]] + pe[idx] : 0
// ---------------------------------------------------------------------------
__global__ __launch_bounds__(256) void embed_kernel(
    const int* __restrict__ pos1, const int* __restrict__ pos2,
    const float* __restrict__ emb,
    float* __restrict__ Xf, bf16_t* __restrict__ Xb)
{
  const int nn = blockIdx.x;   // 0..128
  const int b  = blockIdx.y;   // 0..255
  const int e  = threadIdx.x;  // 0..255
  float v;
  if (nn == 0) {
    v = emb[256 + e];
  } else {
    const int m = nn - 1;
    const int base = b * 128;
    int idx = 0;
    for (int j = 0; j < 128; ++j) {
      const int en = pos1[(base + j) * 2 + 1];
      const int eff = (en > 0) ? en : 129;     // S+1
      idx += (eff <= m) ? 1 : 0;
    }
    const int idxc = idx < 127 ? idx : 127;
    const int st = pos1[(base + idxc) * 2];
    const int en = pos1[(base + idxc) * 2 + 1];
    const bool covered = (idx < 128) && (en > 0) && (m >= st) && (m < en);
    if (covered) {
      const int p2 = pos2[base + idxc];
      const int i2 = e & ~1;
      const float div = __powf(10000.0f, (float)i2 * (1.0f / 256.0f));
      const float ang = (float)idxc / div;
      const float pe  = (e & 1) ? __cosf(ang) : __sinf(ang);
      v = emb[(size_t)p2 * 256 + e] + pe;
    } else {
      v = 0.0f;
    }
  }
  const size_t row = (size_t)b * 129 + nn;
  Xf[row * 256 + e] = v;
  Xb[row * 256 + e] = f2bf(v);
}

__global__ void cvt_bf16(const float* __restrict__ src, bf16_t* __restrict__ dst,
                         size_t nElem)
{
  size_t i = (size_t)blockIdx.x * blockDim.x + threadIdx.x;
  const size_t stride = (size_t)gridDim.x * blockDim.x;
  for (; i < nElem; i += stride) dst[i] = f2bf(src[i]);
}

__global__ __launch_bounds__(256) void disc_kernel(
    const float* __restrict__ bert, const float* __restrict__ Xf,
    const float* __restrict__ dw, const float* __restrict__ db,
    float* __restrict__ out)
{
  const int b = threadIdx.x;  // 0..255
  float acc = db[0];
  const float* bp = bert + (size_t)b * 768;
  for (int i = 0; i < 768; ++i) acc += bp[i] * dw[i];
  const float* pf = Xf + (size_t)b * 129 * 256;   // x[b, 0, :]
  for (int e = 0; e < 256; ++e) acc += pf[e] * dw[768 + e];
  out[b] = 1.0f / (1.0f + __expf(-acc));
}

// ---------------------------------------------------------------------------
extern "C" void kernel_launch(void* const* d_in, const int* in_sizes, int n_in,
                              void* d_out, int out_size, void* d_ws, size_t ws_size,
                              hipStream_t stream)
{
  const int*   pos1   = (const int*)d_in[0];
  const int*   pos2   = (const int*)d_in[1];
  const float* bert   = (const float*)d_in[2];
  const float* emb    = (const float*)d_in[3];
  const float* qkv_w  = (const float*)d_in[4];
  const float* qkv_b  = (const float*)d_in[5];
  const float* out_w  = (const float*)d_in[6];
  const float* out_b  = (const float*)d_in[7];
  const float* ln1_s  = (const float*)d_in[8];
  const float* ln1_b  = (const float*)d_in[9];
  const float* ln2_s  = (const float*)d_in[10];
  const float* ln2_b  = (const float*)d_in[11];
  const float* ff1_w  = (const float*)d_in[12];
  const float* ff1_b  = (const float*)d_in[13];
  const float* ff2_w  = (const float*)d_in[14];
  const float* ff2_b  = (const float*)d_in[15];
  const float* disc_w = (const float*)d_in[16];
  const float* disc_b = (const float*)d_in[17];

  const size_t M = 33024;   // 256 * 129
  char* ws = (char*)d_ws;
  auto alloc = [&](size_t bytes) {
    char* p = ws; ws += (bytes + 255) & ~(size_t)255; return p;
  };
  float*  Xf   = (float*)alloc(M * 256 * 4);
  bf16_t* Xb   = (bf16_t*)alloc(M * 256 * 2);
  bf16_t* QKV  = (bf16_t*)alloc(M * 768 * 2);
  bf16_t* ATT  = (bf16_t*)alloc(M * 256 * 2);
  bf16_t* Hb   = (bf16_t*)alloc(M * 2048 * 2);
  bf16_t* Wqkv = (bf16_t*)alloc((size_t)4 * 768 * 256 * 2);
  bf16_t* Wout = (bf16_t*)alloc((size_t)4 * 256 * 256 * 2);
  bf16_t* Wff1 = (bf16_t*)alloc((size_t)4 * 2048 * 256 * 2);
  bf16_t* Wff2 = (bf16_t*)alloc((size_t)4 * 256 * 2048 * 2);

  cvt_bf16<<<512,  256, 0, stream>>>(qkv_w, Wqkv, (size_t)4 * 768 * 256);
  cvt_bf16<<<256,  256, 0, stream>>>(out_w, Wout, (size_t)4 * 256 * 256);
  cvt_bf16<<<1024, 256, 0, stream>>>(ff1_w, Wff1, (size_t)4 * 2048 * 256);
  cvt_bf16<<<1024, 256, 0, stream>>>(ff2_w, Wff2, (size_t)4 * 256 * 2048);

  embed_kernel<<<dim3(129, 256), 256, 0, stream>>>(pos1, pos2, emb, Xf, Xb);

  for (int l = 0; l < 4; ++l) {
    gemm_kernel<8, 0><<<dim3(6, 258), 256, 0, stream>>>(
        Xb, Wqkv + (size_t)l * 768 * 256, qkv_b + l * 768, 768, 256,
        QKV, nullptr, nullptr, nullptr, nullptr);
    attn_kernel<<<dim3(2, 8, 129), 256, 0, stream>>>(QKV, ATT);
    gemm_kernel<16, 2><<<dim3(1, 258), 256, 0, stream>>>(
        ATT, Wout + (size_t)l * 256 * 256, out_b + l * 256, 256, 256,
        nullptr, ln1_s + l * 256, ln1_b + l * 256, Xf, Xb);
    gemm_kernel<8, 1><<<dim3(16, 258), 256, 0, stream>>>(
        Xb, Wff1 + (size_t)l * 2048 * 256, ff1_b + l * 2048, 2048, 256,
        Hb, nullptr, nullptr, nullptr, nullptr);
    gemm_kernel<16, 2><<<dim3(1, 258), 256, 0, stream>>>(
        Hb, Wff2 + (size_t)l * 256 * 2048, ff2_b + l * 256, 256, 2048,
        nullptr, ln2_s + l * 256, ln2_b + l * 256, Xf, Xb);
  }

  disc_kernel<<<1, 256, 0, stream>>>(bert, Xf, disc_w, disc_b, (float*)d_out);
}
